// CLIPDeMambaDetector_48988396978567
// MI455X (gfx1250) — matrix operand order repaired and checked by reading, not compiled, not run on hardware
//
#include <hip/hip_runtime.h>
#include <hip/hip_bf16.h>
#include <math.h>

// Mamba bidirectional selective scan for MI455X (gfx1250, wave32).
// Pipeline: [WMMA bf16 proj GEMM, x staged bf16 in LDS]
//        -> [WMMA bf16 delta GEMM + softplus]
//        -> [chunked scan phase A, B-slab in LDS] -> [chunk prefix]
//        -> [apply fwd] -> [apply bwd]  (B/C slabs in LDS)

#define DINNER 1024
#define DSTATE 16
#define DTRANK 32
#define BB     4
#define LLEN   2048
#define MROWS  (BB * LLEN)      // 8192
#define NCOLS  128              // 64 fwd (32 dt | 16 B | 16 C) + 64 bwd
#define NCHUNK 16
#define LCHUNK (LLEN / NCHUNK)  // 128

typedef __attribute__((ext_vector_type(16))) __bf16 v16bf;
typedef __attribute__((ext_vector_type(8)))  __bf16 v8bf;
typedef __attribute__((ext_vector_type(2)))  __bf16 v2bf;
typedef __attribute__((ext_vector_type(8)))  float  v8f;
typedef __attribute__((ext_vector_type(4)))  float  v4f;
typedef __attribute__((ext_vector_type(2)))  float  v2f;

__device__ __forceinline__ float softplus_f(float z) {
  return (z > 20.0f) ? z : log1pf(__expf(z));
}

// ---------------------------------------------------------------------------
// Kernel 1: dbc[m][n] = sum_k x[m][k] * W[n][k], n in [0,128) (fwd|bwd weights)
// Block = 256 thr (8 waves). The 16x1024 x tile is converted to bf16 and
// staged in LDS once; each wave owns one 16-col N tile and runs the K loop
// (32 per step -> 32 WMMAs) reading A via ds_load_b128.
// A layout (16x32 bf16): lane m(0-15)/m(16-31): k 0-15 / 16-31 (contiguous)
// B layout (32x16 bf16): lane n / n+16: 16 consecutive k at +0 / +16
// ---------------------------------------------------------------------------
__global__ void k_proj(const float* __restrict__ x,
                       const float* __restrict__ Wf,
                       const float* __restrict__ Wb,
                       float* __restrict__ dbc) {
  __shared__ __bf16 s_x[16][DINNER];          // 32 KB
  const int mt   = blockIdx.x;                // 0..511  (M tile)
  const int tid  = threadIdx.x;
  const int wave = tid >> 5;                  // 0..7    (N tile)
  const int lane = tid & 31;
  const int col  = lane & 15;
  const int half = lane >> 4;

  // Cooperative load + f32->bf16 convert of the x tile (coalesced v2f loads).
  const float* xt = x + mt * 16 * DINNER;
  for (int i = tid; i < 16 * (DINNER / 2); i += 256) {
    const int row = i >> 9;                   // 512 pairs per row
    const int p   = (i & 511) * 2;
    v2f f = *(const v2f*)(xt + row * DINNER + p);
    v2bf pk;
    pk[0] = (__bf16)f[0];
    pk[1] = (__bf16)f[1];
    *(v2bf*)&s_x[row][p] = pk;
  }
  __syncthreads();

  const float* W = (wave < 4) ? (Wf + wave * 16 * DINNER)
                              : (Wb + (wave - 4) * 16 * DINNER);
  const float* wrow = W + col * DINNER;

  v8f acc = {};
  for (int kc = 0; kc < DINNER; kc += 32) {
    const int ab = kc + half * 16;
    v8bf lo = *(const v8bf*)&s_x[col][ab];
    v8bf hi = *(const v8bf*)&s_x[col][ab + 8];
    v16bf a;
#pragma unroll
    for (int i = 0; i < 8; ++i) { a[i] = lo[i]; a[8 + i] = hi[i]; }

    const int bbk = kc + half * 16;
    v4f g0 = *(const v4f*)(wrow + bbk);
    v4f g1 = *(const v4f*)(wrow + bbk + 4);
    v4f g2 = *(const v4f*)(wrow + bbk + 8);
    v4f g3 = *(const v4f*)(wrow + bbk + 12);
    v16bf b;
#pragma unroll
    for (int i = 0; i < 4; ++i) {
      b[i]      = (__bf16)g0[i];
      b[4 + i]  = (__bf16)g1[i];
      b[8 + i]  = (__bf16)g2[i];
      b[12 + i] = (__bf16)g3[i];
    }
    acc = __builtin_amdgcn_wmma_f32_16x16x32_bf16(false, a, false, b,
                                                  (short)0, acc, false, false);
  }
  const int m0 = mt * 16 + half * 8;
#pragma unroll
  for (int r = 0; r < 8; ++r)
    dbc[(m0 + r) * NCOLS + wave * 16 + col] = acc[r];
}

// ---------------------------------------------------------------------------
// Kernel 2: delta[dir][m][d] = softplus(dbc_delta[m] . dt_proj_w[d] + bias[d])
// K = 32 => exactly one WMMA per 16x16 tile.
// ---------------------------------------------------------------------------
__global__ void k_delta(const float* __restrict__ dbc,
                        const float* __restrict__ dtwf,
                        const float* __restrict__ dtbf,
                        const float* __restrict__ dtwb,
                        const float* __restrict__ dtbb,
                        float* __restrict__ delta) {
  const int dir  = blockIdx.y;
  const int tile = blockIdx.x * 8 + (threadIdx.x >> 5);  // 0..32767
  const int mt   = tile >> 6;   // 0..511
  const int nt   = tile & 63;   // 0..63
  const int lane = threadIdx.x & 31;
  const int col  = lane & 15;
  const int half = lane >> 4;
  const int doff = dir * 64;
  const float* dtw = dir ? dtwb : dtwf;
  const float* dtb = dir ? dtbb : dtbf;

  const float* arow = dbc + (mt * 16 + col) * NCOLS + doff;
  const int ab = half * 16;
  v16bf a, b;
#pragma unroll
  for (int i = 0; i < 8; ++i) {
    a[i]     = (__bf16)arow[ab + i];
    a[8 + i] = (__bf16)arow[ab + 8 + i];
  }
  const int d_col = nt * 16 + col;
  const float* wrow = dtw + d_col * DTRANK + half * 16;
#pragma unroll
  for (int i = 0; i < 16; ++i) b[i] = (__bf16)wrow[i];

  v8f acc = {};
  acc = __builtin_amdgcn_wmma_f32_16x16x32_bf16(false, a, false, b,
                                                (short)0, acc, false, false);
  const float bias = dtb[d_col];
  float* out = delta + dir * (MROWS * DINNER);
  const int m0 = mt * 16 + half * 8;
#pragma unroll
  for (int r = 0; r < 8; ++r)
    out[(m0 + r) * DINNER + d_col] = softplus_f(acc[r] + bias);
}

// ---------------------------------------------------------------------------
// Kernel 3 (phase A): per-(dir,b,chunk,d) local scan summary with h=0 start.
// Every thread in a block shares (dir,b,chunk); the chunk's Bm slab
// (128 x 16 f32, 8KB) is staged in LDS and read as broadcast ds loads.
// Block = 256 threads = quarter of the 1024 d channels.
// ---------------------------------------------------------------------------
__global__ void k_chunk(const float* __restrict__ x,
                        const float* __restrict__ dbc,
                        const float* __restrict__ delta,
                        const float* __restrict__ Alog_f,
                        const float* __restrict__ Alog_b,
                        float* __restrict__ chA,
                        float* __restrict__ chB) {
  __shared__ float s_b[LCHUNK][DSTATE];       // 8 KB
  int tb = blockIdx.x >> 2;                   // block-uniform part
  const int d = ((blockIdx.x & 3) << 8) + threadIdx.x;
  const int c = tb & (NCHUNK - 1); tb >>= 4;
  const int b = tb & 3;            tb >>= 2;
  const int dir  = tb;
  const int doff = dir * 64;

  // Stage Bm slab for this chunk (reversal applied here for dir==1).
  {
    const int i  = threadIdx.x;               // exactly LCHUNK*2 threads
    const int j  = i >> 1;
    const int h8 = (i & 1) * 8;
    const int tt = c * LCHUNK + j;
    const int l  = dir ? (LLEN - 1 - tt) : tt;
    const int m  = b * LLEN + l;
    const float* src = dbc + m * NCOLS + doff + DTRANK + h8;
    *(v4f*)&s_b[j][h8]     = *(const v4f*)src;
    *(v4f*)&s_b[j][h8 + 4] = *(const v4f*)(src + 4);
  }
  __syncthreads();

  const float* Alog = dir ? Alog_b : Alog_f;
  float a[DSTATE], prod[DSTATE], h[DSTATE];
#pragma unroll
  for (int s = 0; s < DSTATE; ++s) {
    a[s]    = -__expf(Alog[d * DSTATE + s]);
    prod[s] = 1.0f;
    h[s]    = 0.0f;
  }
  const float* dl = delta + dir * (MROWS * DINNER);
  for (int j = 0; j < LCHUNK; ++j) {
    const int tt = c * LCHUNK + j;
    const int l  = dir ? (LLEN - 1 - tt) : tt;   // bwd dir scans right-to-left
    const int m  = b * LLEN + l;
    {   // prefetch delta/x a few steps ahead (4 KB stride stream)
      const int ttp = (tt + 8 < LLEN) ? (tt + 8) : (LLEN - 1);
      const int lp  = dir ? (LLEN - 1 - ttp) : ttp;
      const int mp  = b * LLEN + lp;
      __builtin_prefetch(dl + mp * DINNER + d, 0, 3);
      __builtin_prefetch(x  + mp * DINNER + d, 0, 3);
    }
    const float de = dl[m * DINNER + d];
    const float xv = x[m * DINNER + d];
    const float du = de * xv;
#pragma unroll
    for (int s = 0; s < DSTATE; ++s) {
      const float dA = __expf(de * a[s]);
      prod[s] *= dA;
      h[s] = dA * h[s] + du * s_b[j][s];
    }
  }
  const int base = (((dir * BB + b) * NCHUNK + c) * DINNER + d) * DSTATE;
#pragma unroll
  for (int s = 0; s < DSTATE; ++s) {
    chA[base + s] = prod[s];
    chB[base + s] = h[s];
  }
}

// ---------------------------------------------------------------------------
// Kernel 4 (phase B): exclusive prefix over the 16 chunk summaries.
// ---------------------------------------------------------------------------
__global__ void k_prefix(const float* __restrict__ chA,
                         const float* __restrict__ chB,
                         float* __restrict__ pref) {
  int t = blockIdx.x * blockDim.x + threadIdx.x;   // 131072 threads
  const int s = t & 15;   t >>= 4;
  const int d = t & 1023; t >>= 10;
  const int b = t & 3;    t >>= 2;
  const int dir = t;
  float p = 0.0f;
  for (int c = 0; c < NCHUNK; ++c) {
    const int idx = (((dir * BB + b) * NCHUNK + c) * DINNER + d) * DSTATE + s;
    pref[idx] = p;
    p = chA[idx] * p + chB[idx];
  }
}

// ---------------------------------------------------------------------------
// Kernel 5 (phase C): replay chunk with carry-in, emit y.
// B and C slabs (128 x 32 f32, 16KB) staged in LDS.
// dir==0 writes y_f + (D + D_b)*x ; dir==1 (launched after) adds y_b.
// ---------------------------------------------------------------------------
__global__ void k_apply(const float* __restrict__ x,
                        const float* __restrict__ dbc,
                        const float* __restrict__ delta,
                        const float* __restrict__ Alog,
                        const float* __restrict__ Dp,
                        const float* __restrict__ Dp2,
                        const float* __restrict__ pref,
                        float* __restrict__ out,
                        const int dir) {
  __shared__ float s_bc[LCHUNK][2 * DSTATE];  // 16 KB
  int tb = blockIdx.x >> 2;
  const int d = ((blockIdx.x & 3) << 8) + threadIdx.x;
  const int c = tb & (NCHUNK - 1); tb >>= 4;
  const int b = tb & 3;
  const int doff = dir * 64;

  {   // stage B|C slab for this chunk
    const int i   = threadIdx.x;              // LCHUNK*2 threads, 16 floats each
    const int j   = i >> 1;
    const int h16 = (i & 1) * 16;
    const int tt  = c * LCHUNK + j;
    const int l   = dir ? (LLEN - 1 - tt) : tt;
    const int m   = b * LLEN + l;
    const float* src = dbc + m * NCOLS + doff + DTRANK + h16;
    *(v4f*)&s_bc[j][h16]      = *(const v4f*)src;
    *(v4f*)&s_bc[j][h16 + 4]  = *(const v4f*)(src + 4);
    *(v4f*)&s_bc[j][h16 + 8]  = *(const v4f*)(src + 8);
    *(v4f*)&s_bc[j][h16 + 12] = *(const v4f*)(src + 12);
  }
  __syncthreads();

  float a[DSTATE], h[DSTATE];
  const int base = (((dir * BB + b) * NCHUNK + c) * DINNER + d) * DSTATE;
#pragma unroll
  for (int s = 0; s < DSTATE; ++s) {
    a[s] = -__expf(Alog[d * DSTATE + s]);
    h[s] = pref[base + s];
  }
  const float skip = dir ? 0.0f : (Dp[d] + Dp2[d]);
  const float* dl = delta + dir * (MROWS * DINNER);
  for (int j = 0; j < LCHUNK; ++j) {
    const int tt = c * LCHUNK + j;
    const int l  = dir ? (LLEN - 1 - tt) : tt;
    const int m  = b * LLEN + l;
    {
      const int ttp = (tt + 8 < LLEN) ? (tt + 8) : (LLEN - 1);
      const int lp  = dir ? (LLEN - 1 - ttp) : ttp;
      const int mp  = b * LLEN + lp;
      __builtin_prefetch(dl + mp * DINNER + d, 0, 3);
      __builtin_prefetch(x  + mp * DINNER + d, 0, 3);
    }
    const float de = dl[m * DINNER + d];
    const float xv = x[m * DINNER + d];
    const float du = de * xv;
    float y = 0.0f;
#pragma unroll
    for (int s = 0; s < DSTATE; ++s) {
      const float dA = __expf(de * a[s]);
      h[s] = dA * h[s] + du * s_bc[j][s];
      y += h[s] * s_bc[j][DSTATE + s];
    }
    float* o = out + m * DINNER + d;
    if (dir == 0) *o = y + skip * xv;
    else          *o += y;
  }
}

// ---------------------------------------------------------------------------
extern "C" void kernel_launch(void* const* d_in, const int* in_sizes, int n_in,
                              void* d_out, int out_size, void* d_ws, size_t ws_size,
                              hipStream_t stream) {
  const float* x     = (const float*)d_in[0];
  const float* xpw   = (const float*)d_in[1];
  const float* dtw   = (const float*)d_in[2];
  const float* dtb   = (const float*)d_in[3];
  const float* Alog  = (const float*)d_in[4];
  const float* Dp    = (const float*)d_in[5];
  const float* xpwb  = (const float*)d_in[6];
  const float* dtwb  = (const float*)d_in[7];
  const float* dtbb  = (const float*)d_in[8];
  const float* Alogb = (const float*)d_in[9];
  const float* Dpb   = (const float*)d_in[10];
  float* out = (float*)d_out;

  // Workspace partition (floats): ~92 MB total
  float* ws    = (float*)d_ws;
  float* dbc   = ws;                                   // 8192*128        =  1,048,576
  float* delta = dbc + MROWS * NCOLS;                  // 2*8192*1024     = 16,777,216
  float* chA   = delta + 2 * MROWS * DINNER;           // 2*4*16*1024*16  =  2,097,152
  float* chB   = chA + 2 * BB * NCHUNK * DINNER * DSTATE;
  float* pref  = chB + 2 * BB * NCHUNK * DINNER * DSTATE;

  k_proj  <<<dim3(MROWS / 16), dim3(256), 0, stream>>>(x, xpw, xpwb, dbc);
  k_delta <<<dim3(4096, 2),    dim3(256), 0, stream>>>(dbc, dtw, dtb, dtwb, dtbb, delta);
  k_chunk <<<dim3(512),        dim3(256), 0, stream>>>(x, dbc, delta, Alog, Alogb, chA, chB);
  k_prefix<<<dim3(512),        dim3(256), 0, stream>>>(chA, chB, pref);
  k_apply <<<dim3(256),        dim3(256), 0, stream>>>(x, dbc, delta, Alog,  Dp,  Dpb, pref, out, 0);
  k_apply <<<dim3(256),        dim3(256), 0, stream>>>(x, dbc, delta, Alogb, Dpb, Dpb, pref, out, 1);
}
